// MyNeuralNetwork_50337016709769
// MI455X (gfx1250) — compile-verified, matrix operand order checked by reading, not run on hardware
//
#include <hip/hip_runtime.h>

// MLP 30->24->19->14->10->6->2->1 over 1M rows, fused, WMMA f16 math / f32 acc.
// Transposed per-layer formulation D[neuron, row] = W x Act^T + bias makes the
// layer-to-layer handoff a pure lane-local ReLU + f32->f16 pack (no LDS).
// Two independent 16-row tiles are processed per loop iteration in straight-line
// code so tile B's VALU packs fill tile A's WMMA hazard slots.

typedef __attribute__((ext_vector_type(16))) _Float16 v16h;
typedef __attribute__((ext_vector_type(8)))  float    v8f;

#define WMMA_F16(a, b, c) \
  __builtin_amdgcn_wmma_f32_16x16x32_f16(false, (a), false, (b), (short)0, (c), false, false)

// A-matrix fragment (16x32 f16, M = neuron, K = fan_in), zero-padded,
// branch-free (clamped addresses + value masks).
__device__ __forceinline__ v16h load_w_frag(const float* __restrict__ W,
                                            int fan_out, int fan_in,
                                            int ntile, int lane) {
  const int   m     = (lane & 15) + 16 * ntile;  // output neuron
  const int   h     = lane >> 4;
  const int   mc    = (m < fan_out) ? m : 0;
  const float mMask = (m < fan_out) ? 1.0f : 0.0f;
  const float* row  = W + (long)mc * fan_in;     // W is [fan_out, fan_in] row-major
  v16h a;
#pragma unroll
  for (int i = 0; i < 8; ++i) {
    const int k0 = 8 * h + i;
    const int k1 = 16 + 8 * h + i;
    const float w0 = row[(k0 < fan_in) ? k0 : 0] * ((k0 < fan_in) ? mMask : 0.0f);
    const float w1 = row[(k1 < fan_in) ? k1 : 0] * ((k1 < fan_in) ? mMask : 0.0f);
    a[i]     = (_Float16)w0;
    a[8 + i] = (_Float16)w1;
  }
  return a;
}

// Bias as WMMA C operand: elem[r] at (M = r + 8h + 16*ntile, N = lane%16).
__device__ __forceinline__ v8f load_b_frag(const float* __restrict__ b,
                                           int fan_out, int ntile, int lane) {
  const int h = lane >> 4;
  v8f c;
#pragma unroll
  for (int r = 0; r < 8; ++r) {
    const int m = r + 8 * h + 16 * ntile;
    c[r] = b[(m < fan_out) ? m : 0] * ((m < fan_out) ? 1.0f : 0.0f);
  }
  return c;
}

// B-matrix fragment for layer 1: B = X^T (K = feature, N = batch row).
// Rows are 120 B (8 B aligned) -> float2 loads. Pair (30,31) is padding:
// clamped address + cndmask, no control flow.
__device__ __forceinline__ v16h load_x_frag(const float* __restrict__ x,
                                            long rowBase, int lane) {
  const int n = lane & 15;
  const int h = lane >> 4;
  const float2* r2 = (const float2*)(x + (rowBase + n) * 30);
  v16h b;
#pragma unroll
  for (int j = 0; j < 4; ++j) {
    const float2 p = r2[4 * h + j];            // features 8h+2j, 8h+2j+1
    b[2 * j]     = (_Float16)p.x;
    b[2 * j + 1] = (_Float16)p.y;
  }
#pragma unroll
  for (int j = 0; j < 4; ++j) {
    const int  pi = 8 + 4 * h + j;             // features 2*pi, 2*pi+1
    const bool ok = pi < 15;                    // pi==15 -> features 30,31 (pad)
    const float2 p = r2[ok ? pi : 8];           // clamped, always in-bounds
    b[8 + 2 * j]     = (_Float16)(ok ? p.x : 0.0f);
    b[8 + 2 * j + 1] = (_Float16)(ok ? p.y : 0.0f);
  }
  return b;
}

// f32->f16 pack first (v_cvt_pk_f16_f32), then ReLU in packed f16
// (v_pk_max_num_f16). relu(cvt(x)) == cvt(relu(x)).
__device__ __forceinline__ v16h pack2(v8f d0, v8f d1) {
  v16h b;
#pragma unroll
  for (int i = 0; i < 8; ++i) {
    b[i]     = (_Float16)d0[i];   // neurons 8h+i
    b[8 + i] = (_Float16)d1[i];   // neurons 16+8h+i
  }
  const v16h z = {};
  return __builtin_elementwise_max(b, z);
}

__device__ __forceinline__ v16h pack1(v8f d0) {
  v16h b = {};                     // neurons >= 16 don't exist -> zero
#pragma unroll
  for (int i = 0; i < 8; ++i) b[i] = (_Float16)d0[i];
  const v16h z = {};
  return __builtin_elementwise_max(b, z);
}

__global__ void __launch_bounds__(256)
mlp7_wmma_kernel(const float* __restrict__ x,
                 const float* __restrict__ W1, const float* __restrict__ b1,
                 const float* __restrict__ W2, const float* __restrict__ b2,
                 const float* __restrict__ W3, const float* __restrict__ b3,
                 const float* __restrict__ W4, const float* __restrict__ b4,
                 const float* __restrict__ W5, const float* __restrict__ b5,
                 const float* __restrict__ W6, const float* __restrict__ b6,
                 const float* __restrict__ W7, const float* __restrict__ b7,
                 float* __restrict__ out, int nTiles) {
  const int lane   = threadIdx.x & 31;
  const int wave   = blockIdx.x * (blockDim.x >> 5) + (threadIdx.x >> 5);
  const int nWaves = gridDim.x * (blockDim.x >> 5);

  // Weight fragments (A operands), built once per wave, live in VGPRs.
  const v16h a1_0 = load_w_frag(W1, 24, 30, 0, lane);
  const v16h a1_1 = load_w_frag(W1, 24, 30, 1, lane);
  const v16h a2_0 = load_w_frag(W2, 19, 24, 0, lane);
  const v16h a2_1 = load_w_frag(W2, 19, 24, 1, lane);
  const v16h a3   = load_w_frag(W3, 14, 19, 0, lane);
  const v16h a4   = load_w_frag(W4, 10, 14, 0, lane);
  const v16h a5   = load_w_frag(W5,  6, 10, 0, lane);
  const v16h a6   = load_w_frag(W6,  2,  6, 0, lane);
  const v16h a7   = load_w_frag(W7,  1,  2, 0, lane);

  // Bias fragments (C operands).
  const v8f c1_0 = load_b_frag(b1, 24, 0, lane);
  const v8f c1_1 = load_b_frag(b1, 24, 1, lane);
  const v8f c2_0 = load_b_frag(b2, 19, 0, lane);
  const v8f c2_1 = load_b_frag(b2, 19, 1, lane);
  const v8f c3   = load_b_frag(b3, 14, 0, lane);
  const v8f c4   = load_b_frag(b4, 10, 0, lane);
  const v8f c5   = load_b_frag(b5,  6, 0, lane);
  const v8f c6   = load_b_frag(b6,  2, 0, lane);
  const v8f c7   = load_b_frag(b7,  1, 0, lane);

  const int nPairs = nTiles >> 1;
  for (int p = wave; p < nPairs; p += nWaves) {
    const long rb0 = (long)(2 * p) << 4;   // 32 batch rows per iteration
    const long rb1 = rb0 + 16;

    v16h hB0 = load_x_frag(x, rb0, lane);
    v16h hB1 = load_x_frag(x, rb1, lane);

    // Layer 1: 30 -> 24 (two neuron tiles each)
    v8f d00 = WMMA_F16(a1_0, hB0, c1_0);
    v8f d01 = WMMA_F16(a1_1, hB0, c1_1);
    v8f d10 = WMMA_F16(a1_0, hB1, c1_0);
    v8f d11 = WMMA_F16(a1_1, hB1, c1_1);
    hB0 = pack2(d00, d01);
    hB1 = pack2(d10, d11);
    // Layer 2: 24 -> 19
    d00 = WMMA_F16(a2_0, hB0, c2_0);
    d01 = WMMA_F16(a2_1, hB0, c2_1);
    d10 = WMMA_F16(a2_0, hB1, c2_0);
    d11 = WMMA_F16(a2_1, hB1, c2_1);
    hB0 = pack2(d00, d01);
    hB1 = pack2(d10, d11);
    // Layers 3..6 with ReLU
    d00 = WMMA_F16(a3, hB0, c3);
    d10 = WMMA_F16(a3, hB1, c3);
    hB0 = pack1(d00); hB1 = pack1(d10);
    d00 = WMMA_F16(a4, hB0, c4);
    d10 = WMMA_F16(a4, hB1, c4);
    hB0 = pack1(d00); hB1 = pack1(d10);
    d00 = WMMA_F16(a5, hB0, c5);
    d10 = WMMA_F16(a5, hB1, c5);
    hB0 = pack1(d00); hB1 = pack1(d10);
    d00 = WMMA_F16(a6, hB0, c6);
    d10 = WMMA_F16(a6, hB1, c6);
    hB0 = pack1(d00); hB1 = pack1(d10);
    // Layer 7: linear, single output neuron (M=0 -> h==0 lanes, elem 0)
    d00 = WMMA_F16(a7, hB0, c7);
    d10 = WMMA_F16(a7, hB1, c7);

    if ((lane >> 4) == 0) {
      out[rb0 + (lane & 15)] = d00[0];
      out[rb1 + (lane & 15)] = d10[0];
    }
  }

  // Tail tile if nTiles is odd (not hit for 1M rows; kept for generality).
  if ((nTiles & 1) && wave == 0) {
    const long rb = (long)(nTiles - 1) << 4;
    v16h hB = load_x_frag(x, rb, lane);
    v8f d0 = WMMA_F16(a1_0, hB, c1_0);
    v8f d1 = WMMA_F16(a1_1, hB, c1_1);
    hB = pack2(d0, d1);
    d0 = WMMA_F16(a2_0, hB, c2_0);
    d1 = WMMA_F16(a2_1, hB, c2_1);
    hB = pack2(d0, d1);
    d0 = WMMA_F16(a3, hB, c3); hB = pack1(d0);
    d0 = WMMA_F16(a4, hB, c4); hB = pack1(d0);
    d0 = WMMA_F16(a5, hB, c5); hB = pack1(d0);
    d0 = WMMA_F16(a6, hB, c6); hB = pack1(d0);
    d0 = WMMA_F16(a7, hB, c7);
    if ((lane >> 4) == 0) out[rb + (lane & 15)] = d0[0];
  }
}

extern "C" void kernel_launch(void* const* d_in, const int* in_sizes, int n_in,
                              void* d_out, int out_size, void* d_ws, size_t ws_size,
                              hipStream_t stream) {
  const float* x  = (const float*)d_in[0];
  const float* W1 = (const float*)d_in[1];  const float* b1 = (const float*)d_in[2];
  const float* W2 = (const float*)d_in[3];  const float* b2 = (const float*)d_in[4];
  const float* W3 = (const float*)d_in[5];  const float* b3 = (const float*)d_in[6];
  const float* W4 = (const float*)d_in[7];  const float* b4 = (const float*)d_in[8];
  const float* W5 = (const float*)d_in[9];  const float* b5 = (const float*)d_in[10];
  const float* W6 = (const float*)d_in[11]; const float* b6 = (const float*)d_in[12];
  const float* W7 = (const float*)d_in[13]; const float* b7 = (const float*)d_in[14];
  float* out = (float*)d_out;

  const int nrows  = in_sizes[0] / 30;   // 1048576
  const int nTiles = nrows / 16;         // 65536 tiles of 16 rows

  const int wavesPerBlock = 8;           // 256 threads = 8 wave32
  int blocks = 1024;                     // 8192 waves -> 4 tile-pairs each
  const int nPairs = nTiles >> 1;
  if (blocks * wavesPerBlock > nPairs && nPairs > 0) {
    blocks = (nPairs + wavesPerBlock - 1) / wavesPerBlock;
  }
  if (blocks < 1) blocks = 1;

  mlp7_wmma_kernel<<<blocks, 256, 0, stream>>>(
      x, W1, b1, W2, b2, W3, b3, W4, b4, W5, b5, W6, b6, W7, b7, out, nTiles);
}
